// MultiScaleRetentionMixin_81432579932326
// MI455X (gfx1250) — matrix operand order repaired and checked by reading, not compiled, not run on hardware
//
#include <hip/hip_runtime.h>
#include <hip/hip_bf16.h>

// ---------------------------------------------------------------------------
// MultiScaleRetention forward for MI455X (gfx1250), bf16 WMMA everywhere.
// B=2, S=2048, Dm=2048, H=16, DK=DV=128, C=64, N=32 chunks.
// ---------------------------------------------------------------------------

typedef __attribute__((ext_vector_type(16))) __bf16 v16bf;
typedef __attribute__((ext_vector_type(8)))  __bf16 v8bf;
typedef __attribute__((ext_vector_type(8)))  float  v8f;

static __device__ inline unsigned short f2bf(float f) {
  unsigned int u = __float_as_uint(f);
  u += 0x7FFFu + ((u >> 16) & 1u);            // round-to-nearest-even
  return (unsigned short)(u >> 16);
}

// A-operand fragment (16-bit A 16x32 layout):
// lane L holds row M=L%16; element e -> K = kbase + (e&8)*2 + (L/16)*8 + (e&7)
static __device__ inline v16bf ldsA(const unsigned short* rp, int kbase, int hf) {
  const v8bf lo = *(const v8bf*)(rp + kbase + hf * 8);
  const v8bf hi = *(const v8bf*)(rp + kbase + 16 + hf * 8);
  return __builtin_shufflevector(lo, hi, 0,1,2,3,4,5,6,7,8,9,10,11,12,13,14,15);
}

// B-operand fragment (16-bit B 32x16 layout), from an LDS tile stored [N][K]:
// lane L holds col N=L%16; element e -> K = kbase + (e&8)*2 + (L/16)*8 + (e&7)
static __device__ inline v16bf ldsB(const unsigned short* rp, int kbase, int hf) {
  const v8bf lo = *(const v8bf*)(rp + kbase + hf * 8);
  const v8bf hi = *(const v8bf*)(rp + kbase + 16 + hf * 8);
  return __builtin_shufflevector(lo, hi, 0,1,2,3,4,5,6,7,8,9,10,11,12,13,14,15);
}

static __device__ inline v8f wmma_bf16(v16bf a, v16bf b, v8f c) {
  return __builtin_amdgcn_wmma_f32_16x16x32_bf16(false, a, false, b, (short)0, c,
                                                 false, false);
}

// ---------------------------------------------------------------------------
// f32 -> bf16 convert (row-major copy)
// ---------------------------------------------------------------------------
__global__ __launch_bounds__(256) void f32_to_bf16_kernel(
    const float* __restrict__ in, unsigned short* __restrict__ out, int n) {
  int i = blockIdx.x * 256 + threadIdx.x;
  if (i < n) out[i] = f2bf(in[i]);
}

// ---------------------------------------------------------------------------
// f32 [R][Ncols] -> bf16 [Ncols][R] (LDS-tiled transpose + convert).
// Used once per weight so the GEMM can stage B with contiguous 16B copies.
// ---------------------------------------------------------------------------
__global__ __launch_bounds__(256) void transpose_f32_to_bf16_kernel(
    const float* __restrict__ in, unsigned short* __restrict__ out,
    int R, int Ncols) {
  __shared__ float tile[32][33];
  const int c  = threadIdx.x & 31;
  const int r0 = threadIdx.x >> 5;           // 0..7
  const int kb = blockIdx.x * 32;            // row base in 'in'
  const int nb = blockIdx.y * 32;            // col base in 'in'
#pragma unroll
  for (int rr = r0; rr < 32; rr += 8)
    tile[rr][c] = in[(size_t)(kb + rr) * Ncols + nb + c];
  __syncthreads();
#pragma unroll
  for (int rr = r0; rr < 32; rr += 8)
    out[(size_t)(nb + rr) * R + kb + c] = f2bf(tile[c][rr]);
}

// ---------------------------------------------------------------------------
// C[M,N] (f32) = A[M,K] (bf16, row-major) * BT[N,K] (bf16, pre-transposed).
// Block tile 128x128, Ktile 32, double-buffered LDS (1 barrier per K-step).
// Staging: all 4 global loads issued before the 4 LDS stores (overlapped
// latencies, regs die before fragment regs go live -> no spills).
// Prefetch of tile k+2 makes the staging wait a WGP$ hit.
// 8 waves: wave = (wm 0..3) x (wn 0..1); each wave: 2x4 tiles of 16x16.
// ---------------------------------------------------------------------------
__global__ __launch_bounds__(256)
__attribute__((amdgpu_waves_per_eu(1)))
void gemm_bf16_kernel(
    const unsigned short* __restrict__ A, const unsigned short* __restrict__ BT,
    float* __restrict__ C, int M, int N, int K) {
  __shared__ unsigned short As[2][128][40];   // [buf][m][k], pad to 40
  __shared__ unsigned short Bs[2][128][40];   // [buf][n][k]

  const int tid  = threadIdx.x;
  const int lane = tid & 31;
  const int wave = tid >> 5;
  const int wm   = wave & 3;
  const int wn   = wave >> 2;
  const int hf   = lane >> 4;
  const int l16  = lane & 15;
  const int bm   = blockIdx.y * 128;
  const int bn   = blockIdx.x * 128;

  v8f acc[2][4];
  const v8f vzero = {0.f,0.f,0.f,0.f,0.f,0.f,0.f,0.f};
#pragma unroll
  for (int mt = 0; mt < 2; ++mt)
#pragma unroll
    for (int nt = 0; nt < 4; ++nt) acc[mt][nt] = vzero;

  int row_[2], kc_[2];
#pragma unroll
  for (int i = 0; i < 2; ++i) {
    int v = tid + i * 256;
    row_[i] = v >> 2;  kc_[i] = (v & 3) * 8;   // 128 rows x 32 k (8 bf16/vec)
  }

  const int NT = K / 32;

  // prologue: stage tile 0 (A and BT use the same contiguous pattern)
  {
    uint4 av[2], bv[2];
#pragma unroll
    for (int i = 0; i < 2; ++i) {
      av[i] = *(const uint4*)(A  + (size_t)(bm + row_[i]) * K + kc_[i]);
      bv[i] = *(const uint4*)(BT + (size_t)(bn + row_[i]) * K + kc_[i]);
    }
#pragma unroll
    for (int i = 0; i < 2; ++i) {
      *(uint4*)(&As[0][row_[i]][kc_[i]]) = av[i];
      *(uint4*)(&Bs[0][row_[i]][kc_[i]]) = bv[i];
    }
  }
  __syncthreads();

  for (int kt = 0; kt < NT; ++kt) {
    const int cur = kt & 1, nxt = cur ^ 1;

    // stage next tile: batch loads, then batch stores; regs die immediately
    if (kt + 1 < NT) {
      int k0 = (kt + 1) * 32;
      uint4 av[2], bv[2];
#pragma unroll
      for (int i = 0; i < 2; ++i) {
        av[i] = *(const uint4*)(A  + (size_t)(bm + row_[i]) * K + k0 + kc_[i]);
        bv[i] = *(const uint4*)(BT + (size_t)(bn + row_[i]) * K + k0 + kc_[i]);
      }
#pragma unroll
      for (int i = 0; i < 2; ++i) {
        *(uint4*)(&As[nxt][row_[i]][kc_[i]]) = av[i];
        *(uint4*)(&Bs[nxt][row_[i]][kc_[i]]) = bv[i];
      }
      if (kt + 2 < NT) {                     // gfx1250 global_prefetch of k+2
        int k2 = (kt + 2) * 32;
        __builtin_prefetch(A  + (size_t)(bm + row_[0]) * K + k2 + kc_[0], 0, 1);
        __builtin_prefetch(BT + (size_t)(bn + row_[0]) * K + k2 + kc_[0], 0, 1);
      }
    }

    // compute on current buffer; B fragments 2 at a time (low pressure)
    v16bf afrag[2];
#pragma unroll
    for (int mt = 0; mt < 2; ++mt)
      afrag[mt] = ldsA(&As[cur][wm * 32 + mt * 16 + l16][0], 0, hf);
#pragma unroll
    for (int g = 0; g < 2; ++g) {
      v16bf bfrag[2];
#pragma unroll
      for (int j = 0; j < 2; ++j)
        bfrag[j] = ldsB(&Bs[cur][wn * 64 + (g * 2 + j) * 16 + l16][0], 0, hf);
#pragma unroll
      for (int mt = 0; mt < 2; ++mt)
#pragma unroll
        for (int j = 0; j < 2; ++j)
          acc[mt][g * 2 + j] = wmma_bf16(afrag[mt], bfrag[j], acc[mt][g * 2 + j]);
    }

    __syncthreads();
  }

#pragma unroll
  for (int mt = 0; mt < 2; ++mt)
#pragma unroll
    for (int nt = 0; nt < 4; ++nt) {
      int col = bn + wn * 64 + nt * 16 + l16;
#pragma unroll
      for (int r = 0; r < 8; ++r) {
        int row = bm + wm * 32 + mt * 16 + r + 8 * hf;
        C[(size_t)row * N + col] = acc[mt][nt][r];
      }
    }
}

// ---------------------------------------------------------------------------
// Chunkwise retention. One block per (b, h, e-slice of 32). 128 blocks.
// 8 waves; wave w owns state tiles (e-tile = w&1, d-tiles = (w>>1)*2..+1)
// of the 32x128 slice of S^T[e][d], kept in registers (2 x v8f).
// ---------------------------------------------------------------------------
__global__ __launch_bounds__(256) void retention_kernel(
    const float* __restrict__ qg, const float* __restrict__ kg,
    const float* __restrict__ vg, float* __restrict__ og) {
  __shared__ unsigned short q_s [64][136];   // [t][d]  (rope'd, *DK^-0.5)
  __shared__ unsigned short k_s [64][136];   // [t][d]  (rope'd)
  __shared__ unsigned short kTd_s[128][72];  // [d][t]  (k * kdec[t])
  __shared__ unsigned short vT_s [32][72];   // [e_loc][t]
  __shared__ unsigned short Am_s [64][72];   // [t][s]  masked q k^T
  __shared__ unsigned short STb_s[32][136];  // [e_loc][d] bf16 state snapshot

  const int tid   = threadIdx.x;
  const int lane  = tid & 31;
  const int wave  = tid >> 5;
  const int hf    = lane >> 4;
  const int l16   = lane & 15;
  const int bh    = blockIdx.x >> 2;         // (b,h)
  const int ebase = (blockIdx.x & 3) * 32;   // e-slice
  const int b     = bh >> 4;
  const int h     = bh & 15;

  const float gamma = 1.0f - exp2f(-5.0f - (float)h);
  const float lg    = log2f(gamma);
  const float gC    = exp2f(64.0f * lg);
  const float qscale = 0.08838834764831845f;        // DK^-0.5
  const float l2_10000_over64 = 13.287712379549449f / 64.0f;

  const int etS  = wave & 1;                 // state: local e-tile
  const int dgrp = wave >> 1;                // state: d-tile group (0..3)
  const int tt_o = wave >> 1;                // output: t-tile
  const int etl_o = wave & 1;                // output: local e-tile

  const v8f vzero = {0.f,0.f,0.f,0.f,0.f,0.f,0.f,0.f};
  v8f S[2];
  S[0] = vzero; S[1] = vzero;

  for (int ch = 0; ch < 32; ++ch) {
    __syncthreads();  // prior-iteration LDS readers done

    // 1) snapshot state slice to LDS (bf16), then decay registers by gamma^C
#pragma unroll
    for (int j = 0; j < 2; ++j) {
      int dt = dgrp * 2 + j;
#pragma unroll
      for (int r = 0; r < 8; ++r) {
        int e = etS * 16 + r + 8 * hf;
        int d = dt * 16 + l16;
        STb_s[e][d] = f2bf(S[j][r]);
        S[j][r] *= gC;
      }
    }

    // 2) stage chunk q,k with RoPE (+ kdec-scaled k^T), and v^T slice
    const int sbase = ch * 64;
    for (int it = tid; it < 4096; it += 256) {       // (t, i<64) pairs
      int t = it >> 6;
      int i = it & 63;
      float inv = exp2f(-(float)i * l2_10000_over64);
      float ang = (float)(sbase + t) * inv;
      float cs = __cosf(ang), sn = __sinf(ang);
      size_t base = ((((size_t)b * 2048 + sbase + t) * 16 + h) << 7);
      float q1 = qg[base + i], q2 = qg[base + i + 64];
      q_s[t][i]      = f2bf((q1 * cs - q2 * sn) * qscale);
      q_s[t][i + 64] = f2bf((q2 * cs + q1 * sn) * qscale);
      float k1 = kg[base + i], k2 = kg[base + i + 64];
      float ko1 = k1 * cs - k2 * sn;
      float ko2 = k2 * cs + k1 * sn;
      k_s[t][i]      = f2bf(ko1);
      k_s[t][i + 64] = f2bf(ko2);
      float kd = exp2f((float)(63 - t) * lg);        // gamma^(C-1-t)
      kTd_s[i][t]      = f2bf(ko1 * kd);
      kTd_s[i + 64][t] = f2bf(ko2 * kd);
    }
    for (int it = tid; it < 2048; it += 256) {       // (t, e_loc<32)
      int t  = it >> 5;
      int el = it & 31;
      size_t base = ((((size_t)b * 2048 + sbase + t) * 16 + h) << 7);
      vT_s[el][t] = f2bf(vg[base + ebase + el]);
    }
    __syncthreads();

    // 3) M1: A[t,s] = sum_d q[t,d] k[s,d], masked by gamma^(t-s) (t>=s)
    {
      int tt = wave & 3;
#pragma unroll
      for (int si = 0; si < 2; ++si) {
        int ss = (wave >> 2) * 2 + si;
        v8f a = vzero;
#pragma unroll
        for (int kd = 0; kd < 4; ++kd) {
          v16bf qa = ldsA(&q_s[tt * 16 + l16][0], kd * 32, hf);
          v16bf kb = ldsB(&k_s[ss * 16 + l16][0], kd * 32, hf);
          a = wmma_bf16(qa, kb, a);
        }
#pragma unroll
        for (int r = 0; r < 8; ++r) {
          int t = tt * 16 + r + 8 * hf;
          int s = ss * 16 + l16;
          float val = (t >= s) ? a[r] * exp2f((float)(t - s) * lg) : 0.0f;
          Am_s[t][s] = f2bf(val);
        }
      }
    }
    __syncthreads();

    // 4) o[t,e] = cross[t]*(q . S_old)[t,e] + (A . v)[t,e]; 1 tile per wave
    {
      v8f a = vzero;
#pragma unroll
      for (int kd = 0; kd < 4; ++kd) {              // K over d (q . S_old)
        v16bf qa = ldsA(&q_s[tt_o * 16 + l16][0], kd * 32, hf);
        v16bf sb = ldsB(&STb_s[etl_o * 16 + l16][0], kd * 32, hf);
        a = wmma_bf16(qa, sb, a);
      }
#pragma unroll
      for (int r = 0; r < 8; ++r) {                 // * gamma^(t+1)
        int t = tt_o * 16 + r + 8 * hf;
        a[r] *= exp2f((float)(t + 1) * lg);
      }
#pragma unroll
      for (int ks = 0; ks < 2; ++ks) {              // K over s (A . v)
        v16bf aa = ldsA(&Am_s[tt_o * 16 + l16][0], ks * 32, hf);
        v16bf vb = ldsB(&vT_s[etl_o * 16 + l16][0], ks * 32, hf);
        a = wmma_bf16(aa, vb, a);
      }
#pragma unroll
      for (int r = 0; r < 8; ++r) {
        int t = tt_o * 16 + r + 8 * hf;
        int e = ebase + etl_o * 16 + l16;
        og[(((size_t)b * 2048 + sbase + t) * 16 + h) * 128 + e] = a[r];
      }
    }

    // 5) state update: S^T[e][d] += sum_t v[t,e] * (k[t,d]*kdec[t])
#pragma unroll
    for (int j = 0; j < 2; ++j) {
      int dt = dgrp * 2 + j;
#pragma unroll
      for (int kt = 0; kt < 2; ++kt) {
        v16bf va = ldsA(&vT_s[etS * 16 + l16][0], kt * 32, hf);
        v16bf kb = ldsB(&kTd_s[dt * 16 + l16][0], kt * 32, hf);
        S[j] = wmma_bf16(va, kb, S[j]);
      }
    }
  }
}

// ---------------------------------------------------------------------------
// per-(b,s,h) gated RMSNorm: rms(o)*w * g*sigmoid(g)  -> bf16
// One wave (32 lanes) per 128-wide row.
// ---------------------------------------------------------------------------
__global__ __launch_bounds__(256) void gated_rmsnorm_kernel(
    const float* __restrict__ o, const float* __restrict__ g,
    const float* __restrict__ w, unsigned short* __restrict__ out, int nrows) {
  int row  = blockIdx.x * 8 + (threadIdx.x >> 5);
  int lane = threadIdx.x & 31;
  if (row >= nrows) return;
  const float* op = o + (size_t)row * 128;
  const float* gp = g + (size_t)row * 128;
  float vals[4], gv[4];
  float ss = 0.f;
#pragma unroll
  for (int j = 0; j < 4; ++j) {
    int e = lane + 32 * j;
    vals[j] = op[e];
    gv[j]   = gp[e];
    ss += vals[j] * vals[j];
  }
#pragma unroll
  for (int off = 16; off >= 1; off >>= 1) ss += __shfl_xor(ss, off, 32);
  float r = rsqrtf(ss * (1.0f / 128.0f) + 1e-5f);
#pragma unroll
  for (int j = 0; j < 4; ++j) {
    int e = lane + 32 * j;
    float gg = gv[j];
    float sw = gg / (1.0f + __expf(-gg));
    out[(size_t)row * 128 + e] = f2bf(vals[j] * r * w[e] * sw);
  }
}

// ---------------------------------------------------------------------------
extern "C" void kernel_launch(void* const* d_in, const int* in_sizes, int n_in,
                              void* d_out, int out_size, void* d_ws, size_t ws_size,
                              hipStream_t stream) {
  (void)in_sizes; (void)n_in; (void)out_size; (void)ws_size;
  const float* x   = (const float*)d_in[0];
  const float* Wq  = (const float*)d_in[1];
  const float* Wk  = (const float*)d_in[2];
  const float* Wv  = (const float*)d_in[3];
  const float* Wg  = (const float*)d_in[4];
  const float* gnw = (const float*)d_in[5];
  const float* Wo  = (const float*)d_in[6];

  const int M = 4096, N = 2048, K = 2048;           // B*S, Dm, Dm
  char* ws = (char*)d_ws;
  size_t off = 0;
  unsigned short* xb  = (unsigned short*)(ws + off); off += (size_t)M * K * 2;
  unsigned short* Wqb = (unsigned short*)(ws + off); off += (size_t)K * N * 2;  // [N][K]
  unsigned short* Wkb = (unsigned short*)(ws + off); off += (size_t)K * N * 2;
  unsigned short* Wvb = (unsigned short*)(ws + off); off += (size_t)K * N * 2;
  unsigned short* Wgb = (unsigned short*)(ws + off); off += (size_t)K * N * 2;
  unsigned short* Wob = (unsigned short*)(ws + off); off += (size_t)K * N * 2;
  float* qf = (float*)(ws + off); off += (size_t)M * N * 4;
  float* kf = (float*)(ws + off); off += (size_t)M * N * 4;
  float* vf = (float*)(ws + off); off += (size_t)M * N * 4;
  float* gf = (float*)(ws + off); off += (size_t)M * N * 4;
  float* of = (float*)(ws + off); off += (size_t)M * N * 4;
  unsigned short* ogb = (unsigned short*)(ws + off); off += (size_t)M * N * 2;

  // activations: plain bf16 convert; weights: transpose + convert -> [N][K]
  f32_to_bf16_kernel<<<(M * K + 255) / 256, 256, 0, stream>>>(x, xb, M * K);
  dim3 tgrid(K / 32, N / 32);
  transpose_f32_to_bf16_kernel<<<tgrid, 256, 0, stream>>>(Wq, Wqb, K, N);
  transpose_f32_to_bf16_kernel<<<tgrid, 256, 0, stream>>>(Wk, Wkb, K, N);
  transpose_f32_to_bf16_kernel<<<tgrid, 256, 0, stream>>>(Wv, Wvb, K, N);
  transpose_f32_to_bf16_kernel<<<tgrid, 256, 0, stream>>>(Wg, Wgb, K, N);
  transpose_f32_to_bf16_kernel<<<tgrid, 256, 0, stream>>>(Wo, Wob, K, N);

  // q,k,v,g projections
  dim3 ggrid(N / 128, M / 128);
  gemm_bf16_kernel<<<ggrid, 256, 0, stream>>>(xb, Wqb, qf, M, N, K);
  gemm_bf16_kernel<<<ggrid, 256, 0, stream>>>(xb, Wkb, kf, M, N, K);
  gemm_bf16_kernel<<<ggrid, 256, 0, stream>>>(xb, Wvb, vf, M, N, K);
  gemm_bf16_kernel<<<ggrid, 256, 0, stream>>>(xb, Wgb, gf, M, N, K);

  // chunkwise retention (RoPE fused into staging), 4-way DV split
  retention_kernel<<<128, 256, 0, stream>>>(qf, kf, vf, of);

  // gated RMSNorm -> bf16
  gated_rmsnorm_kernel<<<(65536 + 7) / 8, 256, 0, stream>>>(of, gf, gnw, ogb, 65536);

  // output projection into d_out (f32)
  gemm_bf16_kernel<<<ggrid, 256, 0, stream>>>(ogb, Wob, (float*)d_out, M, N, K);
}
